// Bottleneck_41884521071061
// MI455X (gfx1250) — compile-verified, matrix operand order checked by reading, not run on hardware
//
#include <hip/hip_runtime.h>
#include <hip/hip_bf16.h>

// Problem constants (from reference): B=32, N=1024, D_IN=1024, D_B=256, K=16384
#define TOK   32768     // B*N
#define DIN   1024
#define DB    256
#define KCB   16384

typedef __attribute__((ext_vector_type(2)))  float  v2f;
typedef __attribute__((ext_vector_type(8)))  float  v8f;
typedef __attribute__((ext_vector_type(16))) __bf16 v16bf;

union BF16x16 {
  v16bf v;
  uint4 q[2];
};

__device__ inline unsigned short f2bf(float f) {
  unsigned u = __builtin_bit_cast(unsigned, f);
  unsigned r = (u + 0x7FFFu + ((u >> 16) & 1u)) >> 16;
  return (unsigned short)r;
}

// ---------------------------------------------------------------------------
// f32 GEMM via V_WMMA_F32_16X16X4_F32, compile-time strides:
//   out[M x N] = A[M x Kd] @ W[Kd x N] + bias[N]
// One wave computes MT vertically-stacked 16x16 C tiles; the strided W-column
// loads are shared across the MT independent accumulator chains. All offsets
// are 32-bit element indices with constant strides -> saddr+voffset loads.
// ---------------------------------------------------------------------------
template<int Kd, int N, int MT>
__launch_bounds__(32)
__global__ void proj_f32_wmma_kernel(const float* __restrict__ A,
                                     const float* __restrict__ W,
                                     const float* __restrict__ bias,
                                     float* __restrict__ outF,
                                     unsigned short* __restrict__ outBF) {
  const int nt   = blockIdx.x;            // N tile
  const int mt   = blockIdx.y;            // group of MT M-tiles
  const int lane = threadIdx.x & 31;
  const int r    = lane & 15;             // A row within tile / D column
  const int kbp  = (lane >> 4) << 1;      // 0 or 2 : K sub-offset (f32 wmma)

  const int m0 = mt * (16 * MT);
  unsigned aoff = (unsigned)(m0 + r) * Kd + (unsigned)kbp;          // elements
  unsigned woff = (unsigned)(nt * 16 + r) + (unsigned)kbp * N;      // elements

  v8f zero = {};
  v8f acc[MT];
  #pragma unroll
  for (int t = 0; t < MT; ++t) acc[t] = zero;

  for (int k0 = 0; k0 < Kd; k0 += 4) {
    v2f b;
    b[0] = W[woff];
    b[1] = W[woff + N];
    woff += 4u * N;
    #pragma unroll
    for (int t = 0; t < MT; ++t) {
      v2f a = *(const v2f*)(A + aoff + (unsigned)(t * 16 * Kd));
      acc[t] = __builtin_amdgcn_wmma_f32_16x16x4_f32(false, a, false, b,
                                                     (short)0, acc[t],
                                                     false, false);
    }
    aoff += 4u;
  }

  const float    bv      = bias[nt * 16 + r];
  const int      rowbase = (lane >> 4) << 3;      // D rows 0..7 or 8..15
  const unsigned col     = (unsigned)(nt * 16 + r);
  #pragma unroll
  for (int t = 0; t < MT; ++t) {
    const unsigned obase = (unsigned)(m0 + t * 16 + rowbase) * N + col;
    #pragma unroll
    for (int v = 0; v < 8; ++v) {
      float val = acc[t][v] + bv;
      unsigned o = obase + (unsigned)(v * N);
      outF[o] = val;
      if (outBF) outBF[o] = f2bf(val);
    }
  }
}

// ---------------------------------------------------------------------------
// Row squared-norms: out[r] = sum_j m[r][j]^2  (row length fixed = DB = 256)
// ---------------------------------------------------------------------------
__launch_bounds__(256)
__global__ void row_norm_kernel(const float* __restrict__ m,
                                float* __restrict__ out, int rows) {
  int r = blockIdx.x * blockDim.x + threadIdx.x;
  if (r >= rows) return;
  const float4* p = (const float4*)(m + (unsigned)r * DB);
  float s = 0.f;
  #pragma unroll 4
  for (int i = 0; i < DB / 4; ++i) {
    float4 v = p[i];
    s += v.x * v.x + v.y * v.y + v.z * v.z + v.w * v.w;
  }
  out[r] = s;
}

// ---------------------------------------------------------------------------
// Fused distance + argmin:
//   d[t][c] = |z_t|^2 + |e_c|^2 - 2 * z_t . e_c   ;  qidx[t] = argmin_c d
// One block (8 waves) per 16-token tile. z tile (bf16, 8KB) staged in LDS and
// reused for all 1024 codebook tiles; dots via v_wmma_f32_16x16x32_bf16.
// B-matrix of zf@embT is emb rows => same contiguous bf16 load pattern as A.
// ---------------------------------------------------------------------------
__launch_bounds__(256)
__global__ void dist_argmin_kernel(const unsigned short* __restrict__ z_bf,
                                   const unsigned short* __restrict__ emb_bf,
                                   const float* __restrict__ z_norm,
                                   const float* __restrict__ emb_norm,
                                   int* __restrict__ qidx) {
  __shared__ __align__(16) unsigned short zt[16 * DB];  // 8 KB
  __shared__ float zn[16];
  __shared__ float swv[8][16];
  __shared__ int   swi[8][16];

  const int mt   = blockIdx.x;
  const int tid  = threadIdx.x;
  const int lane = tid & 31;
  const int wid  = tid >> 5;

  // stage 16 x 256 bf16 z tile into LDS (512 uint4 total)
  {
    const uint4* gsrc = (const uint4*)(z_bf + (unsigned)mt * (16 * DB));
    uint4* ldst = (uint4*)zt;
    ldst[tid]       = gsrc[tid];
    ldst[tid + 256] = gsrc[tid + 256];
    if (tid < 16) zn[tid] = z_norm[mt * 16 + tid];
  }
  __syncthreads();

  const int r      = lane & 15;          // A row / B column within tile
  const int kb     = (lane >> 4) << 3;   // 0 or 8 : bf16 wmma K sub-offset
  const int rowoff = (lane >> 4) << 3;   // D rows 0..7 or 8..15

  // preload A (fixed per block): 8 K-chunks of 32 bf16
  v16bf A[8];
  #pragma unroll
  for (int kc = 0; kc < 8; ++kc) {
    const unsigned short* p = &zt[r * DB + kc * 32 + kb];
    BF16x16 t;
    t.q[0] = *(const uint4*)(p);
    t.q[1] = *(const uint4*)(p + 16);
    A[kc] = t.v;
  }
  float znr[8];
  #pragma unroll
  for (int v = 0; v < 8; ++v) znr[v] = zn[rowoff + v];

  float bestv[8];
  int   besti[8];
  #pragma unroll
  for (int v = 0; v < 8; ++v) { bestv[v] = __builtin_inff(); besti[v] = 0; }

  // each wave covers codebook tiles nt = wid, wid+8, ... (32-bit offsets)
  unsigned       eoff  = (unsigned)(wid * 16 + r) * DB;
  const unsigned estep = 8u * 16u * DB;
  int            code  = wid * 16 + r;

  for (int nt = wid; nt < KCB / 16; nt += 8) {
    const unsigned short* erow = emb_bf + eoff;
    __builtin_prefetch(emb_bf + eoff + estep, 0, 1);   // next codebook tile
    v8f acc = {};
    #pragma unroll
    for (int kc = 0; kc < 8; ++kc) {
      const unsigned short* p = erow + kc * 32 + kb;
      BF16x16 t;
      t.q[0] = *(const uint4*)(p);
      t.q[1] = *(const uint4*)(p + 16);
      acc = __builtin_amdgcn_wmma_f32_16x16x32_bf16(false, A[kc], false, t.v,
                                                    (short)0, acc, false, false);
    }
    const float en = emb_norm[(unsigned)code];
    #pragma unroll
    for (int v = 0; v < 8; ++v) {
      float d = znr[v] + en - 2.0f * acc[v];
      if (d < bestv[v]) { bestv[v] = d; besti[v] = code; }
    }
    eoff += estep;
    code += 8 * 16;
  }

  // reduce (min, idx) across the 16 lanes sharing each row (xor masks < 16
  // stay within the half-wave handling those rows)
  #pragma unroll
  for (int v = 0; v < 8; ++v) {
    float bv = bestv[v]; int bi = besti[v];
    #pragma unroll
    for (int m = 8; m >= 1; m >>= 1) {
      float ov = __shfl_xor(bv, m, 32);
      int   oi = __shfl_xor(bi, m, 32);
      if (ov < bv || (ov == bv && oi < bi)) { bv = ov; bi = oi; }
    }
    bestv[v] = bv; besti[v] = bi;
  }
  if ((lane & 15) == 0) {
    #pragma unroll
    for (int v = 0; v < 8; ++v) {
      swv[wid][rowoff + v] = bestv[v];
      swi[wid][rowoff + v] = besti[v];
    }
  }
  __syncthreads();

  // cross-wave reduce, 16 rows
  if (tid < 16) {
    float bv = swv[0][tid]; int bi = swi[0][tid];
    #pragma unroll
    for (int w = 1; w < 8; ++w) {
      float ov = swv[w][tid]; int oi = swi[w][tid];
      if (ov < bv || (ov == bv && oi < bi)) { bv = ov; bi = oi; }
    }
    qidx[mt * 16 + tid] = bi;
  }
}

// ---------------------------------------------------------------------------
// Gather: reg_z[t] = emb[qidx[t]] ; rep[t] = (float)qidx[t]
// one block (64 threads) per token: 64 float4 = 256 floats
// ---------------------------------------------------------------------------
__launch_bounds__(64)
__global__ void gather_kernel(const int* __restrict__ qidx,
                              const float* __restrict__ emb,
                              float* __restrict__ regz,
                              float* __restrict__ rep) {
  const int t = blockIdx.x;
  const int l = threadIdx.x;
  const int idx = qidx[t];
  const float4* src = (const float4*)(emb + (unsigned)idx * DB);
  float4* dst = (float4*)(regz + (unsigned)t * DB);
  dst[l] = src[l];
  if (l == 0) rep[t] = (float)idx;
}

// ---------------------------------------------------------------------------
extern "C" void kernel_launch(void* const* d_in, const int* in_sizes, int n_in,
                              void* d_out, int out_size, void* d_ws, size_t ws_size,
                              hipStream_t stream) {
  (void)in_sizes; (void)n_in; (void)out_size; (void)ws_size;

  const float* x        = (const float*)d_in[0];  // [TOK, DIN]
  const float* W_in     = (const float*)d_in[1];  // [DIN, DB]
  const float* b_in     = (const float*)d_in[2];  // [DB]
  const float* W_out    = (const float*)d_in[3];  // [DB, DIN]
  const float* b_out    = (const float*)d_in[4];  // [DIN]
  const float* codebook = (const float*)d_in[5];  // [KCB, DB]
  const float* W_proj   = (const float*)d_in[6];  // [DB, DB]
  const float* b_proj   = (const float*)d_in[7];  // [DB]

  // outputs, flat in return order: x_hat, rep, z, emb, reg_z
  float* out    = (float*)d_out;
  float* x_hat  = out;                                      // TOK*DIN
  float* rep    = x_hat + (size_t)TOK * DIN;                // TOK
  float* z      = rep + TOK;                                // TOK*DB
  float* emb    = z + (size_t)TOK * DB;                     // KCB*DB
  float* regz   = emb + (size_t)KCB * DB;                   // TOK*DB

  // workspace layout (16B aligned chunks)
  char* w = (char*)d_ws;
  unsigned short* z_bf   = (unsigned short*)w;                          // 16 MB
  unsigned short* emb_bf = (unsigned short*)(w + (size_t)TOK * DB * 2); // 8 MB
  float* z_norm   = (float*)(w + (size_t)TOK * DB * 2 + (size_t)KCB * DB * 2);
  float* emb_norm = z_norm + TOK;
  int*   qidx     = (int*)(emb_norm + KCB);

  // 1) z = x @ W_in + b_in   (f32 wmma), + bf16 copy
  proj_f32_wmma_kernel<DIN, DB, 4><<<dim3(DB / 16, TOK / 64), 32, 0, stream>>>(
      x, W_in, b_in, z, z_bf);

  // 2) emb = codebook @ W_proj + b_proj (f32 wmma), + bf16 copy
  proj_f32_wmma_kernel<DB, DB, 4><<<dim3(DB / 16, KCB / 64), 32, 0, stream>>>(
      codebook, W_proj, b_proj, emb, emb_bf);

  // 3) squared row norms
  row_norm_kernel<<<TOK / 256, 256, 0, stream>>>(z, z_norm, TOK);
  row_norm_kernel<<<KCB / 256, 256, 0, stream>>>(emb, emb_norm, KCB);

  // 4) fused distance + argmin (bf16 wmma, 275 GFLOP path)
  dist_argmin_kernel<<<TOK / 16, 256, 0, stream>>>(
      z_bf, emb_bf, z_norm, emb_norm, qidx);

  // 5) gather quantized rows -> reg_z, write bottleneck_rep
  gather_kernel<<<TOK, 64, 0, stream>>>(qidx, emb, regz, rep);

  // 6) x_hat = reg_z @ W_out + b_out (f32 wmma)
  proj_f32_wmma_kernel<DB, DIN, 4><<<dim3(DIN / 16, TOK / 64), 32, 0, stream>>>(
      regz, W_out, b_out, x_hat, nullptr);
}
